// TransformerTranslator_22127671509256
// MI455X (gfx1250) — matrix operand order repaired and checked by reading, not compiled, not run on hardware
//
#include <hip/hip_runtime.h>
#include <hip/hip_bf16.h>

// ---------------------------------------------------------------------------
// TransformerTranslator forward for MI455X (gfx1250, wave32, WMMA bf16).
// GEMMs: v_wmma_f32_16x16x32_bf16, 2x2 register-blocked tiles per wave
// (4 WMMAs per 6 vector loads), WGP-scope global_prefetch_b8 (inline asm) of
// the next k-step, non-temporal stores for the 704MB final output.
// Softmax/LN in f32 VALU.
// ---------------------------------------------------------------------------

#define NB   512
#define TT   43
#define HH   128
#define MM   (NB * TT)      // 22016 rows = 1376 M-tiles = 688 M-tile-pairs
#define DKK  96
#define DFFN 2048
#define OUTN 8000

typedef __bf16 bf16_t;
typedef __attribute__((ext_vector_type(16))) __bf16 v16bf;
typedef __attribute__((ext_vector_type(8)))  __bf16 v8bf;
typedef __attribute__((ext_vector_type(8)))  float  v8f;

__device__ __forceinline__ unsigned short f2bf(float f) {
  // round-to-nearest-even f32 -> bf16 (bit pattern)
  unsigned u = __float_as_uint(f);
  u += 0x7FFFu + ((u >> 16) & 1u);
  return (unsigned short)(u >> 16);
}

// WGP-scope prefetch: ISA scope 0 pulls into all cache levels (incl. WGP$),
// unlike __builtin_prefetch which only reaches SE/SYS scope (GL2 only).
__device__ __forceinline__ void prefetch_wgp(const void* p) {
  asm volatile("global_prefetch_b8 %0, off" :: "v"(p));
}

// ---------------------------------------------------------------------------
// Embedding: x[row,h] = emb[tok,h] + pos[t,h]; write f32 + bf16 mirrors.
// ---------------------------------------------------------------------------
__global__ void embed_kernel(const int* __restrict__ inputs,
                             const float* __restrict__ emb,
                             const float* __restrict__ pos,
                             float* __restrict__ xf,
                             bf16_t* __restrict__ xb) {
  int row = blockIdx.x;                 // n*TT + t
  int t   = row % TT;
  int h   = threadIdx.x;                // 128
  int tok = inputs[row];
  float v = emb[(size_t)tok * HH + h] + pos[t * HH + h];
  size_t idx = (size_t)row * HH + h;
  xf[idx] = v;
  ((unsigned short*)xb)[idx] = f2bf(v);
}

// ---------------------------------------------------------------------------
// f32 -> bf16 conversion (for weights)
// ---------------------------------------------------------------------------
__global__ void cvt_bf16_kernel(const float* __restrict__ in,
                                bf16_t* __restrict__ out, int n) {
  int i = blockIdx.x * 256 + threadIdx.x;
  if (i < n) ((unsigned short*)out)[i] = f2bf(in[i]);
}

// ---------------------------------------------------------------------------
// Fragment load helpers (wave32; A/B use this code's contiguous k order, not the ISA table order; D as in ISA):
//   A: lane m & lane 16+m hold row m; half 0 -> K 0..15, half 1 ->
//      K 16..31   (two 16B loads per lane per k-step)
//   B: lane n / 16+n hold column n; half 0 -> K 0..15, half 1 -> K 16..31
//      (one 32B load per lane per k-step)
//   D: VGPR r: lanes 0..15 row r, lanes 16..31 row 8+r; col = lane&15
// ---------------------------------------------------------------------------
__device__ __forceinline__ v16bf load_a_frag(const bf16_t* rowp, int kb, int half) {
  v8bf a0 = *(const v8bf*)(rowp + kb + half * 16);
  v8bf a1 = *(const v8bf*)(rowp + kb + half * 16 + 8);
  v16bf af;
#pragma unroll
  for (int i = 0; i < 8; ++i) { af[i] = a0[i]; af[8 + i] = a1[i]; }
  return af;
}

template<bool RELU, bool OUT_F32, bool OUT_BF16, bool NT>
__device__ __forceinline__ void store_tile(const v8f& acc, float bv, int rowBase,
                                           int col, int ldc,
                                           float* __restrict__ Cf,
                                           bf16_t* __restrict__ Cb) {
#pragma unroll
  for (int r = 0; r < 8; ++r) {
    float v = acc[r] + bv;
    if (RELU) v = fmaxf(v, 0.f);
    size_t idx = (size_t)(rowBase + r) * ldc + col;
    if (OUT_F32) {
      if (NT) __builtin_nontemporal_store(v, &Cf[idx]);   // TH=NT: don't pollute L2
      else    Cf[idx] = v;
    }
    if (OUT_BF16) ((unsigned short*)Cb)[idx] = f2bf(v);
  }
}

// ---------------------------------------------------------------------------
// WMMA GEMM: C[M, ldc](+cofs) = A[M,K](bf16 row-major) @ B[Nn,K](bf16)^T +bias
// Each wave computes a 32x32 block (2x2 WMMA tiles): A/B fragments reused 2x,
// 4 v_wmma per k-step of 32. 8 waves/block. K % 32 == 0, M/Nn % 32 == 0.
// ---------------------------------------------------------------------------
template<bool RELU, bool OUT_F32, bool OUT_BF16, bool NT>
__global__ void wmma_gemm2x2(const bf16_t* __restrict__ A,
                             const bf16_t* __restrict__ B,
                             const float*  __restrict__ bias,
                             float*  __restrict__ Cf,
                             bf16_t* __restrict__ Cb,
                             int Nn, int K, int ldc, int cofs) {
  int lane = threadIdx.x & 31;
  int wave = threadIdx.x >> 5;
  int m0 = blockIdx.x * 2;                    // M tile pair
  int n0 = (blockIdx.y * 8 + wave) * 2;       // N tile pair
  if (n0 * 16 >= Nn) return;                  // wave-uniform: EXEC all-ones

  int half = lane >> 4;
  int l16  = lane & 15;
  const bf16_t* Ap0 = A + (size_t)(m0 * 16 + l16) * K;
  const bf16_t* Ap1 = Ap0 + (size_t)16 * K;
  const bf16_t* Bp0 = B + (size_t)(n0 * 16 + l16) * K;
  const bf16_t* Bp1 = Bp0 + (size_t)16 * K;

  v8f acc00 = {0.f,0.f,0.f,0.f,0.f,0.f,0.f,0.f};
  v8f acc01 = acc00, acc10 = acc00, acc11 = acc00;

#pragma unroll 2
  for (int kb = 0; kb < K; kb += 32) {
    if (kb + 32 < K) {           // prefetch next k-step into WGP$ (scope 0)
      prefetch_wgp(Ap0 + kb + 32);
      prefetch_wgp(Ap1 + kb + 32);
      prefetch_wgp(Bp0 + kb + 32);
      prefetch_wgp(Bp1 + kb + 32);
    }
    v16bf af0 = load_a_frag(Ap0, kb, half);
    v16bf af1 = load_a_frag(Ap1, kb, half);
    v16bf bf0 = *(const v16bf*)(Bp0 + kb + half * 16);
    v16bf bf1 = *(const v16bf*)(Bp1 + kb + half * 16);
    acc00 = __builtin_amdgcn_wmma_f32_16x16x32_bf16(false, af0, false, bf0,
                                                    (short)0, acc00, false, false);
    acc01 = __builtin_amdgcn_wmma_f32_16x16x32_bf16(false, af0, false, bf1,
                                                    (short)0, acc01, false, false);
    acc10 = __builtin_amdgcn_wmma_f32_16x16x32_bf16(false, af1, false, bf0,
                                                    (short)0, acc10, false, false);
    acc11 = __builtin_amdgcn_wmma_f32_16x16x32_bf16(false, af1, false, bf1,
                                                    (short)0, acc11, false, false);
  }

  float bv0 = bias[n0 * 16 + l16];
  float bv1 = bias[n0 * 16 + 16 + l16];
  int col0 = n0 * 16 + l16 + cofs;
  int col1 = col0 + 16;
  int row0 = m0 * 16 + half * 8;
  int row1 = row0 + 16;
  store_tile<RELU, OUT_F32, OUT_BF16, NT>(acc00, bv0, row0, col0, ldc, Cf, Cb);
  store_tile<RELU, OUT_F32, OUT_BF16, NT>(acc01, bv1, row0, col1, ldc, Cf, Cb);
  store_tile<RELU, OUT_F32, OUT_BF16, NT>(acc10, bv0, row1, col0, ldc, Cf, Cb);
  store_tile<RELU, OUT_F32, OUT_BF16, NT>(acc11, bv1, row1, col1, ldc, Cf, Cb);
}

// ---------------------------------------------------------------------------
// Attention: one block per (sequence, head). K/V/Q staged in LDS.
// Wave-per-row softmax with shfl reductions (wave32). Writes bf16 heads
// buffer [M, 192] (head h -> cols h*96 .. h*96+95).
// qkv layout: [M, 576] cols: k1|v1|q1|k2|v2|q2 (96 each).
// ---------------------------------------------------------------------------
__global__ void attn_kernel(const float* __restrict__ qkv,
                            bf16_t* __restrict__ heads) {
  __shared__ float sk[TT * DKK];
  __shared__ float sv[TT * DKK];
  __shared__ float sq[TT * DKK];
  __shared__ float sattn[8 * 64];

  int n = blockIdx.x, head = blockIdx.y;
  int tid = threadIdx.x;
  const float scale = 0.1020620726159658f;   // 1/sqrt(96)
  size_t base = (size_t)(n * TT) * 576 + head * 288;

  for (int i = tid; i < TT * DKK; i += 256) {
    int s = i / DKK, d = i - s * DKK;
    size_t rb = base + (size_t)s * 576 + d;
    sk[i] = qkv[rb];
    sv[i] = qkv[rb + 96];
    sq[i] = qkv[rb + 192];
  }
  __syncthreads();

  int wave = tid >> 5, lane = tid & 31;
  for (int t = wave; t < TT; t += 8) {
    // scores for s = lane and s = lane+32
    float sc0, sc1 = -3.0e38f;
    {
      float acc = 0.f;
      for (int d = 0; d < DKK; ++d) acc += sq[t * DKK + d] * sk[lane * DKK + d];
      sc0 = acc * scale;
      if (lane + 32 < TT) {
        float a2 = 0.f;
        for (int d = 0; d < DKK; ++d)
          a2 += sq[t * DKK + d] * sk[(lane + 32) * DKK + d];
        sc1 = a2 * scale;
      }
    }
    float m = fmaxf(sc0, sc1);
    for (int off = 16; off; off >>= 1) m = fmaxf(m, __shfl_xor(m, off, 32));
    float e0 = __expf(sc0 - m);
    float e1 = (lane + 32 < TT) ? __expf(sc1 - m) : 0.f;
    float sum = e0 + e1;
    for (int off = 16; off; off >>= 1) sum += __shfl_xor(sum, off, 32);
    float inv = 1.f / sum;
    sattn[wave * 64 + lane]      = e0 * inv;   // wave-synchronous LDS use
    sattn[wave * 64 + 32 + lane] = e1 * inv;

    // out[t, d0..d0+2] = sum_s attn[s] * v[s, d]
    int d0 = lane * 3;
    float o0 = 0.f, o1 = 0.f, o2 = 0.f;
    for (int s = 0; s < TT; ++s) {
      float a = sattn[wave * 64 + s];
      o0 += a * sv[s * DKK + d0];
      o1 += a * sv[s * DKK + d0 + 1];
      o2 += a * sv[s * DKK + d0 + 2];
    }
    size_t ob = (size_t)(n * TT + t) * 192 + head * 96 + d0;
    unsigned short* hp = (unsigned short*)heads;
    hp[ob]     = f2bf(o0);
    hp[ob + 1] = f2bf(o1);
    hp[ob + 2] = f2bf(o2);
  }
}

// ---------------------------------------------------------------------------
// Residual add + LayerNorm, wave-per-row (H=128 -> 4 elems/lane on wave32).
// ---------------------------------------------------------------------------
__global__ void add_ln_kernel(const float* __restrict__ a,
                              const float* __restrict__ b,
                              const float* __restrict__ g,
                              const float* __restrict__ be,
                              float* __restrict__ outf,
                              bf16_t* __restrict__ outb) {
  int row  = blockIdx.x * 8 + (threadIdx.x >> 5);   // MM % 8 == 0
  int lane = threadIdx.x & 31;
  size_t rb = (size_t)row * HH;
  float v[4];
  float s = 0.f;
#pragma unroll
  for (int i = 0; i < 4; ++i) {
    int h = lane * 4 + i;
    v[i] = a[rb + h] + b[rb + h];
    s += v[i];
  }
  for (int off = 16; off; off >>= 1) s += __shfl_xor(s, off, 32);
  float mean = s * (1.f / 128.f);
  float var = 0.f;
#pragma unroll
  for (int i = 0; i < 4; ++i) { float d = v[i] - mean; var += d * d; }
  for (int off = 16; off; off >>= 1) var += __shfl_xor(var, off, 32);
  float inv = rsqrtf(var * (1.f / 128.f) + 1e-5f);
#pragma unroll
  for (int i = 0; i < 4; ++i) {
    int h = lane * 4 + i;
    float o = (v[i] - mean) * inv * g[h] + be[h];
    outf[rb + h] = o;
    ((unsigned short*)outb)[rb + h] = f2bf(o);
  }
}

// ---------------------------------------------------------------------------
extern "C" void kernel_launch(void* const* d_in, const int* in_sizes, int n_in,
                              void* d_out, int out_size, void* d_ws, size_t ws_size,
                              hipStream_t stream) {
  (void)in_sizes; (void)n_in; (void)out_size; (void)ws_size;
  const int*   inputs = (const int*)  d_in[0];
  const float* emb    = (const float*)d_in[1];
  const float* pos    = (const float*)d_in[2];
  const float* Wk1 = (const float*)d_in[3];  const float* bk1 = (const float*)d_in[4];
  const float* Wv1 = (const float*)d_in[5];  const float* bv1 = (const float*)d_in[6];
  const float* Wq1 = (const float*)d_in[7];  const float* bq1 = (const float*)d_in[8];
  const float* Wk2 = (const float*)d_in[9];  const float* bk2 = (const float*)d_in[10];
  const float* Wv2 = (const float*)d_in[11]; const float* bv2 = (const float*)d_in[12];
  const float* Wq2 = (const float*)d_in[13]; const float* bq2 = (const float*)d_in[14];
  const float* Wproj = (const float*)d_in[15]; const float* bproj = (const float*)d_in[16];
  const float* g_mh  = (const float*)d_in[17]; const float* b_mh  = (const float*)d_in[18];
  const float* Wff1  = (const float*)d_in[19]; const float* bff1  = (const float*)d_in[20];
  const float* Wff2  = (const float*)d_in[21]; const float* bff2  = (const float*)d_in[22];
  const float* g_ff  = (const float*)d_in[23]; const float* b_ff  = (const float*)d_in[24];
  const float* Wfin  = (const float*)d_in[25]; const float* bfin  = (const float*)d_in[26];

  char* ws = (char*)d_ws;
  size_t off = 0;
  auto carve = [&](size_t bytes) -> char* {
    char* p = ws + off;
    off += (bytes + 255) & ~(size_t)255;
    return p;
  };
  float*  x_f32 = (float*) carve((size_t)MM * HH * 4);
  bf16_t* x_b   = (bf16_t*)carve((size_t)MM * HH * 2);
  float*  qkv   = (float*) carve((size_t)MM * 576 * 4);
  bf16_t* heads = (bf16_t*)carve((size_t)MM * 192 * 2);
  float*  projf = (float*) carve((size_t)MM * HH * 4);
  float*  x1f   = (float*) carve((size_t)MM * HH * 4);
  bf16_t* x1b   = (bf16_t*)carve((size_t)MM * HH * 2);
  bf16_t* ffn1b = (bf16_t*)carve((size_t)MM * DFFN * 2);
  float*  ff2f  = (float*) carve((size_t)MM * HH * 4);
  float*  x2f   = (float*) carve((size_t)MM * HH * 4);
  bf16_t* x2b   = (bf16_t*)carve((size_t)MM * HH * 2);
  bf16_t* wqkvb[6];
  const float* Wqkv[6] = {Wk1, Wv1, Wq1, Wk2, Wv2, Wq2};
  const float* Bqkv[6] = {bk1, bv1, bq1, bk2, bv2, bq2};
  for (int i = 0; i < 6; ++i) wqkvb[i] = (bf16_t*)carve((size_t)DKK * HH * 2);
  bf16_t* wprojb = (bf16_t*)carve((size_t)HH * 192 * 2);
  bf16_t* wff1b  = (bf16_t*)carve((size_t)DFFN * HH * 2);
  bf16_t* wff2b  = (bf16_t*)carve((size_t)HH * DFFN * 2);
  bf16_t* wfinb  = (bf16_t*)carve((size_t)OUTN * HH * 2);

  // 1) embedding (+ bf16 mirror)
  embed_kernel<<<MM, HH, 0, stream>>>(inputs, emb, pos, x_f32, x_b);

  // 2) weight conversions f32 -> bf16
  auto cvt = [&](const float* in, bf16_t* out, int n) {
    cvt_bf16_kernel<<<(n + 255) / 256, 256, 0, stream>>>(in, out, n);
  };
  for (int i = 0; i < 6; ++i) cvt(Wqkv[i], wqkvb[i], DKK * HH);
  cvt(Wproj, wprojb, HH * 192);
  cvt(Wff1,  wff1b,  DFFN * HH);
  cvt(Wff2,  wff2b,  HH * DFFN);
  cvt(Wfin,  wfinb,  OUTN * HH);

  dim3 blk(256);
  const int MP = MM / 32;   // 688 M-tile pairs

  // 3) QKV projections into fused [M,576] buffer (cols: k1|v1|q1|k2|v2|q2)
  for (int i = 0; i < 6; ++i) {
    wmma_gemm2x2<false, true, false, false><<<dim3(MP, 1), blk, 0, stream>>>(
        x_b, wqkvb[i], Bqkv[i], qkv, (bf16_t*)nullptr,
        /*Nn=*/DKK, /*K=*/HH, /*ldc=*/576, /*cofs=*/i * 96);
  }

  // 4) attention -> heads bf16 [M,192]
  attn_kernel<<<dim3(NB, 2), blk, 0, stream>>>(qkv, heads);

  // 5) output projection + residual + LN
  wmma_gemm2x2<false, true, false, false><<<dim3(MP, 1), blk, 0, stream>>>(
      heads, wprojb, bproj, projf, (bf16_t*)nullptr, HH, 192, HH, 0);
  add_ln_kernel<<<MM / 8, blk, 0, stream>>>(projf, x_f32, g_mh, b_mh, x1f, x1b);

  // 6) FFN: relu(x1 @ W1^T + b1) @ W2^T + b2, residual + LN
  wmma_gemm2x2<true, false, true, false><<<dim3(MP, DFFN / 256), blk, 0, stream>>>(
      x1b, wff1b, bff1, (float*)nullptr, ffn1b, DFFN, HH, DFFN, 0);
  wmma_gemm2x2<false, true, false, false><<<dim3(MP, 1), blk, 0, stream>>>(
      ffn1b, wff2b, bff2, ff2f, (bf16_t*)nullptr, HH, DFFN, HH, 0);
  add_ln_kernel<<<MM / 8, blk, 0, stream>>>(ff2f, x1f, g_ff, b_ff, x2f, x2b);

  // 7) final vocab projection -> d_out [M, 8000] f32 (250 N-pairs -> 32 blk.y)
  //    Non-temporal stores: 704MB stream must not evict L2-resident A/W.
  wmma_gemm2x2<false, true, false, true><<<dim3(MP, 32), blk, 0, stream>>>(
      x2b, wfinb, bfin, (float*)d_out, (bf16_t*)nullptr, OUTN, HH, OUTN, 0);
}
